// BayesianLinear_89799176225396
// MI455X (gfx1250) — matrix operand order repaired and checked by reading, not compiled, not run on hardware
//
#include <hip/hip_runtime.h>

typedef __attribute__((ext_vector_type(16))) _Float16 v16h;
typedef __attribute__((ext_vector_type(8)))  _Float16 v8h;
typedef __attribute__((ext_vector_type(4)))  _Float16 v4h;
typedef __attribute__((ext_vector_type(8)))  float    v8f;
typedef __attribute__((ext_vector_type(4)))  float    v4f;

#define IN_F   4096
#define OUT_F  4096
#define BATCH  1024

#define BM 128   // output rows per block (batch dim)
#define BN 128   // output cols per block (feature dim)
#define BK 32    // K per WMMA step
#define NK (IN_F / BK)
#define LDA 40   // halves per LDS row (80 B): 16B-aligned frags, conflict-free banks
#define ABUF (BM * LDA)   // halves per A buffer
#define BBUF (BN * LDA)   // halves per B buffer

__global__ __launch_bounds__(256)
void bayes_linear_wmma(const float* __restrict__ x,
                       const float* __restrict__ eps_w,
                       const float* __restrict__ eps_b,
                       const float* __restrict__ mu_w,
                       const float* __restrict__ log_sigma_w,
                       const float* __restrict__ mu_b,
                       const float* __restrict__ log_sigma_b,
                       float* __restrict__ out)
{
    __shared__ _Float16 As[2 * ABUF];   // double-buffered x tile, f16   (20 KB)
    __shared__ _Float16 Bs[2 * BBUF];   // double-buffered sampled W, f16 (20 KB)

    const int t    = threadIdx.x;
    const int lane = t & 31;
    const int wave = t >> 5;      // 0..7
    const int wm   = wave >> 2;   // 0..1  (M direction)
    const int wn   = wave & 3;    // 0..3  (N direction)

    const int mBlock = blockIdx.y * BM;
    const int nBlock = blockIdx.x * BN;

    v8f acc[4][2];
    #pragma unroll
    for (int mi = 0; mi < 4; ++mi)
        #pragma unroll
        for (int nj = 0; nj < 2; ++nj)
            acc[mi][nj] = (v8f){0.f,0.f,0.f,0.f,0.f,0.f,0.f,0.f};

    const int colLane = lane & 15;
    const int hiHalf  = lane >> 4;        // 0 for lanes 0-15, 1 for lanes 16-31
    const int kbA     = hiHalf * 16;      // A frag: K base 0 or 16
    const int kbB     = hiHalf * 16;      // B frag: K base 0 or 16

    // Per-thread cooperative-load geometry (1024 float4 per tile, 4 per thread)
    int    rowq[4], kqv[4];
    size_t xoff[4], woff[4];
    #pragma unroll
    for (int i = 0; i < 4; ++i) {
        int q   = i * 256 + t;
        rowq[i] = q >> 3;
        kqv[i]  = (q & 7) << 2;
        xoff[i] = (size_t)(mBlock + rowq[i]) * IN_F + kqv[i];
        woff[i] = (size_t)(nBlock + rowq[i]) * IN_F + kqv[i];
    }

    // -------- pipeline prologue: load K-step 0 into registers --------
    v4f xr[4], mr[4], sr[4], er[4];
    #pragma unroll
    for (int i = 0; i < 4; ++i) {
        xr[i] = *(const v4f*)(x           + xoff[i]);
        mr[i] = *(const v4f*)(mu_w        + woff[i]);
        sr[i] = *(const v4f*)(log_sigma_w + woff[i]);
        er[i] = *(const v4f*)(eps_w       + woff[i]);
    }

    int buf = 0;
    for (int kt = 0; kt < NK; ++kt) {
        _Float16* Asb = As + buf * ABUF;
        _Float16* Bsb = Bs + buf * BBUF;

        // ---- convert + store current K-step tiles to LDS[buf] ----
        #pragma unroll
        for (int i = 0; i < 4; ++i) {
            v4h hv;
            hv.x = (_Float16)xr[i].x; hv.y = (_Float16)xr[i].y;
            hv.z = (_Float16)xr[i].z; hv.w = (_Float16)xr[i].w;
            *(v4h*)(Asb + rowq[i] * LDA + kqv[i]) = hv;

            v4h wv;   // fused reparameterization: W = mu + exp(log_sigma) * eps
            wv.x = (_Float16)(mr[i].x + __expf(sr[i].x) * er[i].x);
            wv.y = (_Float16)(mr[i].y + __expf(sr[i].y) * er[i].y);
            wv.z = (_Float16)(mr[i].z + __expf(sr[i].z) * er[i].z);
            wv.w = (_Float16)(mr[i].w + __expf(sr[i].w) * er[i].w);
            *(v4h*)(Bsb + rowq[i] * LDA + kqv[i]) = wv;
        }

        // ---- issue next K-step global loads (overlap with barrier + WMMA) ----
        if (kt + 1 < NK) {
            size_t ko = (size_t)(kt + 1) * BK;
            #pragma unroll
            for (int i = 0; i < 4; ++i) {
                xr[i] = *(const v4f*)(x           + xoff[i] + ko);
                mr[i] = *(const v4f*)(mu_w        + woff[i] + ko);
                sr[i] = *(const v4f*)(log_sigma_w + woff[i] + ko);
                er[i] = *(const v4f*)(eps_w       + woff[i] + ko);
            }
        }

        // ---- prefetch K-step kt+2 (covers pipeline depth) ----
        if (kt + 2 < NK) {
            size_t ko = (size_t)(kt + 2) * BK;
            __builtin_prefetch(x           + xoff[0] + ko, 0, 3);
            __builtin_prefetch(mu_w        + woff[0] + ko, 0, 3);
            __builtin_prefetch(log_sigma_w + woff[0] + ko, 0, 3);
            __builtin_prefetch(eps_w       + woff[0] + ko, 0, 3);
        }

        __syncthreads();   // publish LDS[buf]; also fences reuse of LDS[buf^1]

        // ---- fragment loads, contiguous k order (k = 16*half + s) for A and B ----
        v16h a[4], b[2];
        #pragma unroll
        for (int mi = 0; mi < 4; ++mi) {
            const _Float16* p = Asb + (wm * 64 + mi * 16 + colLane) * LDA;
            v8h lo = *(const v8h*)(p + kbA);        // K = kb .. kb+7
            v8h hi = *(const v8h*)(p + kbA + 8);    // K = kb+8 .. kb+15
            a[mi] = __builtin_shufflevector(lo, hi, 0,1,2,3,4,5,6,7,8,9,10,11,12,13,14,15);
        }
        #pragma unroll
        for (int nj = 0; nj < 2; ++nj) {
            const _Float16* p = Bsb + (wn * 32 + nj * 16 + colLane) * LDA + kbB;
            v8h lo = *(const v8h*)(p);              // K = kbB .. kbB+7
            v8h hi = *(const v8h*)(p + 8);          // K = kbB+8 .. kbB+15
            b[nj] = __builtin_shufflevector(lo, hi, 0,1,2,3,4,5,6,7,8,9,10,11,12,13,14,15);
        }

        // ---- 8 WMMAs per K-step ----
        #pragma unroll
        for (int mi = 0; mi < 4; ++mi)
            #pragma unroll
            for (int nj = 0; nj < 2; ++nj)
                acc[mi][nj] = __builtin_amdgcn_wmma_f32_16x16x32_f16(
                    /*neg_a=*/false, a[mi], /*neg_b=*/false, b[nj],
                    /*c_mod=*/(short)0, acc[mi][nj],
                    /*reuse_a=*/false, /*reuse_b=*/false);

        buf ^= 1;
    }

    // ---- epilogue: sampled bias + fp32 store ----
    const int rowAdd = hiHalf << 3;   // 0 or 8 per C/D layout
    #pragma unroll
    for (int nj = 0; nj < 2; ++nj) {
        int col = nBlock + wn * 32 + nj * 16 + colLane;
        float bias = mu_b[col] + __expf(log_sigma_b[col]) * eps_b[col];
        #pragma unroll
        for (int mi = 0; mi < 4; ++mi) {
            int rbase = mBlock + wm * 64 + mi * 16 + rowAdd;
            #pragma unroll
            for (int r = 0; r < 8; ++r)
                out[(size_t)(rbase + r) * OUT_F + col] = acc[mi][nj][r] + bias;
        }
    }
}

extern "C" void kernel_launch(void* const* d_in, const int* in_sizes, int n_in,
                              void* d_out, int out_size, void* d_ws, size_t ws_size,
                              hipStream_t stream) {
    const float* x           = (const float*)d_in[0];
    const float* eps_w       = (const float*)d_in[1];
    const float* eps_b       = (const float*)d_in[2];
    const float* mu_w        = (const float*)d_in[3];
    const float* log_sigma_w = (const float*)d_in[4];
    const float* mu_b        = (const float*)d_in[5];
    const float* log_sigma_b = (const float*)d_in[6];
    float* out = (float*)d_out;

    dim3 grid(OUT_F / BN, BATCH / BM);   // 32 x 8 = 256 blocks
    bayes_linear_wmma<<<grid, 256, 0, stream>>>(
        x, eps_w, eps_b, mu_w, log_sigma_w, mu_b, log_sigma_b, out);
}
